// QConv1D_26980984554179
// MI455X (gfx1250) — 8-bit matrix operand order repaired and checked by reading, compiled, not run on hardware
//
#include <hip/hip_runtime.h>
#include <hip/hip_bf16.h>

typedef int   v8i __attribute__((ext_vector_type(8)));
typedef float v4f __attribute__((ext_vector_type(4)));

#define NX 1600
#define NF 6400
#define MROWS 8192
#define KTILES 25
#define MTILES 512
#define NTILES 400

#define WS_AMAX   0
#define WS_RATIO  16
#define WS_BQ     256
#define WS_APACK  32768
#define WS_BPACK  (32768 + 13107200)

__global__ void k_absmax(const float* __restrict__ w, unsigned* __restrict__ amax, int n) {
    __shared__ float red[256];
    float m = 0.0f;
    for (int i = blockIdx.x * blockDim.x + threadIdx.x; i < n; i += gridDim.x * blockDim.x)
        m = fmaxf(m, fabsf(w[i]));
    red[threadIdx.x] = m;
    __syncthreads();
    for (int s = 128; s > 0; s >>= 1) {
        if (threadIdx.x < (unsigned)s) red[threadIdx.x] = fmaxf(red[threadIdx.x], red[threadIdx.x + s]);
        __syncthreads();
    }
    if (threadIdx.x == 0) atomicMax(amax, __float_as_uint(red[0]));
}

__global__ void k_bias(const float* __restrict__ bias, const float* __restrict__ in_scale,
                       const float* __restrict__ out_scale, const unsigned* __restrict__ amax,
                       int* __restrict__ bq, float* __restrict__ ratio) {
    int i = blockIdx.x * blockDim.x + threadIdx.x;
    float w_scale = __uint_as_float(*amax) * (1.0f / 127.0f);
    float bscale  = (*in_scale) * w_scale;
    if (i < NF) bq[i] = (int)rintf(bias[i] / bscale);
    if (i == 0) *ratio = bscale / (*out_scale);
}

__global__ void k_wpack(const float* __restrict__ w, const unsigned* __restrict__ amax,
                        unsigned* __restrict__ bpack) {
    int tid  = blockIdx.x * 256 + threadIdx.x;
    int v    = tid & 7;
    int lane = (tid >> 3) & 31;
    int tile = tid >> 8;
    int kt   = tile % KTILES;
    int nt   = tile / KTILES;
    int n    = nt * 16 + (lane & 15);
    int hi   = lane >> 4;
    int kbase = kt * 64 + (v >> 2) * 32 + hi * 16 + (v & 3) * 4;
    float inv_ws = 127.0f / __uint_as_float(*amax);
    unsigned packed = 0;
#pragma unroll
    for (int b = 0; b < 4; ++b) {
        float wv = __builtin_nontemporal_load(&w[(kbase + b) * NF + n]);
        int q = (int)rintf(wv * inv_ws);
        q = max(-128, min(127, q));
        packed |= ((unsigned)(q & 0xFF)) << (8 * b);
    }
    bpack[tid] = packed;
}

__global__ void k_xpack(const float* __restrict__ x, const float* __restrict__ in_scale,
                        unsigned* __restrict__ apack) {
    int tid  = blockIdx.x * 256 + threadIdx.x;
    int v    = tid & 7;
    int lane = (tid >> 3) & 31;
    int tile = tid >> 8;
    int kt   = tile % KTILES;
    int mt   = tile / KTILES;
    int m    = mt * 16 + (lane & 15);
    int hi   = lane >> 4;
    int kbase = kt * 64 + (v >> 2) * 32 + hi * 16 + (v & 3) * 4;
    float inv = 1.0f / (*in_scale);
    const v4f f = __builtin_nontemporal_load((const v4f*)(x + (size_t)m * NX + kbase));
    unsigned packed = 0;
#pragma unroll
    for (int b = 0; b < 4; ++b) {
        int q = (int)rintf(f[b] * inv);
        q = max(-128, min(127, q));
        packed |= ((unsigned)(q & 0xFF)) << (8 * b);
    }
    apack[tid] = packed;
}

__global__ void __launch_bounds__(256, 1)
k_gemm(const v8i* __restrict__ apack, const v8i* __restrict__ bpack,
       const int* __restrict__ bq, const float* __restrict__ ratio_p,
       const float* __restrict__ out_scale_p, float* __restrict__ out) {
    const int lane = threadIdx.x & 31;
    const int wave = threadIdx.x >> 5;
    const int gw   = blockIdx.x * 8 + wave;
    const int mb   = gw % (MTILES / 4);
    const int nb   = gw / (MTILES / 4);
    const int mt0  = mb * 4;
    const int nt0  = nb * 4;

    const v8i* aptr[4];
    const v8i* bptr[4];
#pragma unroll
    for (int i = 0; i < 4; ++i) aptr[i] = apack + (size_t)(mt0 + i) * (KTILES * 32) + lane;
#pragma unroll
    for (int j = 0; j < 4; ++j) bptr[j] = bpack + (size_t)(nt0 + j) * (KTILES * 32) + lane;

    v8i acc[4][4];
#pragma unroll
    for (int i = 0; i < 4; ++i)
#pragma unroll
        for (int j = 0; j < 4; ++j)
            acc[i][j] = (v8i){0, 0, 0, 0, 0, 0, 0, 0};

#pragma unroll
    for (int kt = 0; kt < KTILES; ++kt) {
        v8i af[4], bf[4];
#pragma unroll
        for (int i = 0; i < 4; ++i) af[i] = aptr[i][kt * 32];
#pragma unroll
        for (int j = 0; j < 4; ++j) bf[j] = bptr[j][kt * 32];
#pragma unroll
        for (int i = 0; i < 4; ++i)
#pragma unroll
            for (int j = 0; j < 4; ++j)
                acc[i][j] = __builtin_amdgcn_wmma_i32_16x16x64_iu8(
                     true, af[i],  true, bf[j],
                    acc[i][j],  false,  false);
    }

    const float ratio = *ratio_p;
    const float osc   = *out_scale_p;
    const int nlo = lane & 15;
    const int hi  = lane >> 4;

    int bqv[4];
    const int* bqbase = bq + nt0 * 16 + nlo;
#pragma unroll
    for (int j = 0; j < 4; ++j) bqv[j] = bqbase[j * 16];

    float* obase = out + (size_t)(mt0 * 16 + hi * 8) * NF + nt0 * 16 + nlo;
#pragma unroll
    for (int i = 0; i < 4; ++i) {
#pragma unroll
        for (int r = 0; r < 8; ++r) {
            float* orow = obase + (size_t)(i * 16 + r) * NF;
#pragma unroll
            for (int j = 0; j < 4; ++j) {
                float f = (float)(acc[i][j][r] + bqv[j]) * ratio;
                float q = fminf(127.0f, fmaxf(-128.0f, rintf(f)));
                __builtin_nontemporal_store(q * osc, orow + j * 16);
            }
        }
    }
}

extern "C" void kernel_launch(void* const* d_in, const int* in_sizes, int n_in,
                              void* d_out, int out_size, void* d_ws, size_t ws_size,
                              hipStream_t stream) {
    const float* x         = (const float*)d_in[0];
    const float* weight    = (const float*)d_in[1];
    const float* bias      = (const float*)d_in[2];
    const float* in_scale  = (const float*)d_in[3];
    const float* out_scale = (const float*)d_in[4];
    float* out = (float*)d_out;

    char* ws = (char*)d_ws;
    unsigned* amax  = (unsigned*)(ws + WS_AMAX);
    float*    ratio = (float*)(ws + WS_RATIO);
    int*      bq    = (int*)(ws + WS_BQ);
    unsigned* apack = (unsigned*)(ws + WS_APACK);
    unsigned* bpack = (unsigned*)(ws + WS_BPACK);

    (void)hipMemsetAsync(amax, 0, sizeof(unsigned), stream);

    k_absmax<<<2048, 256, 0, stream>>>(weight, amax, NX * NF);
    k_bias<<<NF / 256, 256, 0, stream>>>(bias, in_scale, out_scale, amax, bq, ratio);
    k_wpack<<<(NX * NF / 4) / 256, 256, 0, stream>>>(weight, amax, bpack);
    k_xpack<<<(MROWS * NX / 4) / 256, 256, 0, stream>>>(x, in_scale, apack);
    k_gemm<<<(MTILES / 4) * (NTILES / 4) / 8, 256, 0, stream>>>(
        (const v8i*)apack, (const v8i*)bpack, bq, ratio, out_scale, out);
}
